// Temporal_Alignment_49237505081659
// MI455X (gfx1250) — matrix operand order repaired and checked by reading, not compiled, not run on hardware
//
#include <hip/hip_runtime.h>
#include <math.h>

typedef _Float16 half_t;
typedef _Float16 v16h __attribute__((ext_vector_type(16)));
typedef _Float16 v8h  __attribute__((ext_vector_type(8)));
typedef float    v8f  __attribute__((ext_vector_type(8)));

#define BB    2
#define CC    64
#define HH    64
#define WWD   64
#define HW    4096
#define OFFC  166
#define MP    176      // 166 padded to multiple of 16
#define K33   1152     // 128*9 (multiple of 32)
#define K44R  1494     // 166*9
#define K44P  1504     // padded to multiple of 32
#define KPW   192

// ---------------------------------------------------------------- pool (2x2 avg of concat(x,y))
__global__ void pool_kernel(const float* __restrict__ x, const float* __restrict__ y,
                            float* __restrict__ pooled) {
    int tid = blockIdx.x * blockDim.x + threadIdx.x;
    if (tid >= BB * 128 * 1024) return;
    int p = tid & 1023;
    int c = (tid >> 10) & 127;
    int b = tid >> 17;
    int py = p >> 5, px = p & 31;
    const float* src = (c < 64) ? x : y;
    int cc = c & 63;
    const float* base = src + (((size_t)b * 64 + cc) * 64 + py * 2) * 64 + px * 2;
    pooled[tid] = 0.25f * (base[0] + base[1] + base[64] + base[65]);
}

// ---------------------------------------------------------------- pack fp32 weights -> fp16 padded MxK
__global__ void pack_w_kernel(const float* __restrict__ src, half_t* __restrict__ dst,
                              int Mreal, int Kreal, int Mp, int Kp) {
    int tid = blockIdx.x * blockDim.x + threadIdx.x;
    if (tid >= Mp * Kp) return;
    int k = tid % Kp, m = tid / Kp;
    half_t v = (half_t)0.0f;
    if (m < Mreal && k < Kreal) v = (half_t)src[(size_t)m * Kreal + k];
    dst[tid] = v;
}

// ---------------------------------------------------------------- im2col (3x3, pad=1), transposed N x Kp, fp16
__global__ void im2col_kernel(const float* __restrict__ src0, const float* __restrict__ src1,
                              int C0, int C1, int Hd, int Wd,
                              int Kreal, int Kp, int Nn, half_t* __restrict__ dst) {
    int tid = blockIdx.x * blockDim.x + threadIdx.x;
    if (tid >= Nn * Kp) return;
    int k = tid % Kp;
    int n = tid / Kp;
    half_t v = (half_t)0.0f;
    if (k < Kreal) {
        int ci = k / 9, kk = k % 9;
        int hw = Hd * Wd;
        int b = n / hw, p = n % hw;
        int yy = p / Wd + kk / 3 - 1;
        int xx = p % Wd + kk % 3 - 1;
        if (yy >= 0 && yy < Hd && xx >= 0 && xx < Wd) {
            const float* s;
            size_t idx;
            if (ci < C0) { s = src0; idx = (((size_t)b * C0 + ci) * Hd + yy) * Wd + xx; }
            else         { s = src1; idx = (((size_t)b * C1 + (ci - C0)) * Hd + yy) * Wd + xx; }
            v = (half_t)s[idx];
        }
    }
    dst[tid] = v;
}

// ---------------------------------------------------------------- fragment loader (two b128 -> v16h)
__device__ __forceinline__ v16h load_frag(const half_t* __restrict__ p, int hiOff) {
    v8h lo = *(const v8h*)p;
    v8h hi = *(const v8h*)(p + hiOff);
    return __builtin_shufflevector(lo, hi, 0,1,2,3,4,5,6,7,8,9,10,11,12,13,14,15);
}

// ---------------------------------------------------------------- WMMA GEMM: C[MxN] = A[MpxKp] * BT[NxKp]^T
// Each wave: one 16-row M tile x four 16-col N tiles; A fragment reused across 4 WMMAs.
// Register double-buffering across K-steps: next step's fragments are in flight
// while the current step's 4 WMMAs execute (loads have no consumer until the
// next iteration, so waits land after the WMMA burst).
// epi: 0 = plain, 1 = * sigmoid(gate at half-res), 2 = + bias
__global__ void __launch_bounds__(256)
wmma_gemm_kernel(const half_t* __restrict__ A, const half_t* __restrict__ BT, int Kp,
                 float* __restrict__ out, int Mreal, int HWo, int epi,
                 const float* __restrict__ gate, const float* __restrict__ bias, int gateC) {
    int wave = threadIdx.x >> 5;
    int lane = threadIdx.x & 31;
    int n0 = (blockIdx.x * 8 + wave) * 64;
    int m0 = blockIdx.y * 16;
    int hsel  = lane >> 4;         // 0: lanes 0-15, 1: lanes 16-31
    int lm    = lane & 15;
    int klo   = hsel * 8;          // A fragment K sub-offset (ISA 16-bit A 16x32 layout)
    int koffB = hsel * 8;          // B fragment K sub-offset (ISA 16-bit B 32x16 layout)

    const half_t* aRow = A + (size_t)(m0 + lm) * Kp;
    const half_t* bCol[4];
#pragma unroll
    for (int t = 0; t < 4; ++t)
        bCol[t] = BT + (size_t)(n0 + t * 16 + lm) * Kp;

    v8f acc[4] = {};

    // prologue: load first fragments
    v16h a_cur = load_frag(aRow + klo, 16);
    v16h b_cur[4];
#pragma unroll
    for (int t = 0; t < 4; ++t) b_cur[t] = load_frag(bCol[t] + koffB, 16);

#pragma unroll 2
    for (int k0 = 0; k0 < Kp; k0 += 32) {
        // issue next step's loads first (clamped on last iter; value unused)
        int k1 = (k0 + 32 < Kp) ? (k0 + 32) : 0;
        v16h a_nxt = load_frag(aRow + k1 + klo, 16);
        v16h b_nxt[4];
#pragma unroll
        for (int t = 0; t < 4; ++t) b_nxt[t] = load_frag(bCol[t] + k1 + koffB, 16);

        // WMMA burst on current fragments (independent of in-flight loads)
#pragma unroll
        for (int t = 0; t < 4; ++t)
            acc[t] = __builtin_amdgcn_wmma_f32_16x16x32_f16(
                false, a_cur, false, b_cur[t], (short)0, acc[t], false, false);

        a_cur = a_nxt;
#pragma unroll
        for (int t = 0; t < 4; ++t) b_cur[t] = b_nxt[t];
    }

#pragma unroll
    for (int t = 0; t < 4; ++t) {
        int nn = n0 + t * 16 + lm;
        int bimg = nn / HWo;
        int p = nn % HWo;
#pragma unroll
        for (int r = 0; r < 8; ++r) {
            int mm = m0 + r + hsel * 8;
            if (mm >= Mreal) continue;
            float v = acc[t][r];
            if (epi == 1) {
                int yy = p >> 6, xx = p & 63;            // HWo==4096 path only
                float g = gate[(((size_t)bimg * gateC + mm) * 32 + (yy >> 1)) * 32 + (xx >> 1)];
                v *= 1.0f / (1.0f + __expf(-g));
            } else if (epi == 2) {
                v += bias[mm];
            }
            out[((size_t)bimg * Mreal + mm) * HWo + p] = v;
        }
    }
}

// ---------------------------------------------------------------- deformable conv (grouped, bilinear) + ReLU
__device__ __forceinline__ float dsamp(const float* __restrict__ ch, int yy, int xx) {
    if (yy < 0 || yy >= HH || xx < 0 || xx >= WWD) return 0.0f;
    return ch[yy * WWD + xx];
}

__global__ void deform_kernel(const float* __restrict__ x, const float* __restrict__ offs,
                              const float* __restrict__ w, const float* __restrict__ bias,
                              int K, int pad, int offBase, int catBase,
                              half_t* __restrict__ catBT) {
    int tid = blockIdx.x * blockDim.x + threadIdx.x;
    if (tid >= BB * CC * HW) return;
    int p  = tid & 4095;
    int co = (tid >> 12) & 63;
    int b  = tid >> 18;
    int g  = co >> 3;                 // 8 groups, 8 in-ch per group
    int yr = p >> 6, xr = p & 63;
    int K2 = K * K;
    float acc = bias[co];
    const float* xb = x + (size_t)b * CC * HW;
    const float* ob = offs + (size_t)b * OFFC * HW;
    for (int kk = 0; kk < K2; ++kk) {
        float oy = ob[(size_t)(offBase + 2 * kk    ) * HW + p];
        float ox = ob[(size_t)(offBase + 2 * kk + 1) * HW + p];
        float py = (float)yr + (float)(kk / K - pad) + oy;
        float px = (float)xr + (float)(kk % K - pad) + ox;
        float fy = floorf(py), fx = floorf(px);
        int y0 = (int)fy, x0 = (int)fx;
        float dy = py - fy, dx = px - fx;
        float w00 = (1.0f - dy) * (1.0f - dx);
        float w01 = (1.0f - dy) * dx;
        float w10 = dy * (1.0f - dx);
        float w11 = dy * dx;
        for (int ci = 0; ci < 8; ++ci) {
            const float* ch = xb + (size_t)(g * 8 + ci) * HW;
            float s = dsamp(ch, y0, x0)         * w00
                    + dsamp(ch, y0, x0 + 1)     * w01
                    + dsamp(ch, y0 + 1, x0)     * w10
                    + dsamp(ch, y0 + 1, x0 + 1) * w11;
            acc = fmaf(s, w[(size_t)(co * 8 + ci) * K2 + kk], acc);
        }
    }
    float r = fmaxf(acc, 0.0f);
    // write straight into transposed (pixel-major, K-contiguous) fp16 buffer for the pointwise GEMM
    catBT[((size_t)b * HW + p) * KPW + catBase + co] = (half_t)r;
}

// ---------------------------------------------------------------- host
extern "C" void kernel_launch(void* const* d_in, const int* in_sizes, int n_in,
                              void* d_out, int out_size, void* d_ws, size_t ws_size,
                              hipStream_t stream) {
    const float* x    = (const float*)d_in[0];
    const float* y    = (const float*)d_in[1];
    const float* k2_w = (const float*)d_in[2];
    const float* k3_w = (const float*)d_in[3];
    const float* k4_w = (const float*)d_in[4];
    const float* w3   = (const float*)d_in[5];
    const float* b3   = (const float*)d_in[6];
    const float* w5   = (const float*)d_in[7];
    const float* b5   = (const float*)d_in[8];
    const float* w7   = (const float*)d_in[9];
    const float* b7   = (const float*)d_in[10];
    const float* pw_w = (const float*)d_in[11];
    const float* pw_b = (const float*)d_in[12];
    float* out = (float*)d_out;

    char* wp = (char*)d_ws;
    auto alloc = [&](size_t bytes) { char* p = wp; wp += (bytes + 255) & ~(size_t)255; return p; };

    float*  pooled = (float*) alloc((size_t)BB * 128 * 1024 * 4);
    half_t* wA2    = (half_t*)alloc((size_t)MP * K33 * 2);
    half_t* wA3    = (half_t*)alloc((size_t)MP * K33 * 2);
    half_t* wA4    = (half_t*)alloc((size_t)MP * K44P * 2);
    half_t* wApw   = (half_t*)alloc((size_t)64 * KPW * 2);
    half_t* btPool = (half_t*)alloc((size_t)2048 * K33 * 2);
    float*  o2     = (float*) alloc((size_t)BB * OFFC * 1024 * 4);
    half_t* btXY   = (half_t*)alloc((size_t)8192 * K33 * 2);
    float*  outg   = (float*) alloc((size_t)BB * OFFC * HW * 4);
    half_t* btOut  = (half_t*)alloc((size_t)8192 * K44P * 2);
    float*  offs   = (float*) alloc((size_t)BB * OFFC * HW * 4);
    half_t* catBT  = (half_t*)alloc((size_t)8192 * KPW * 2);

    // 1. pooled concat(x,y)
    pool_kernel<<<(BB * 128 * 1024 + 255) / 256, 256, 0, stream>>>(x, y, pooled);

    // 2. pack weights to fp16 padded
    pack_w_kernel<<<(MP * K33 + 255) / 256, 256, 0, stream>>>(k2_w, wA2, OFFC, K33, MP, K33);
    pack_w_kernel<<<(MP * K33 + 255) / 256, 256, 0, stream>>>(k3_w, wA3, OFFC, K33, MP, K33);
    pack_w_kernel<<<(MP * K44P + 255) / 256, 256, 0, stream>>>(k4_w, wA4, OFFC, K44R, MP, K44P);
    pack_w_kernel<<<(64 * KPW + 255) / 256, 256, 0, stream>>>(pw_w, wApw, 64, KPW, 64, KPW);

    // 3. conv2(pooled) -> o2 (gate logits at 32x32)
    im2col_kernel<<<(2048 * K33 + 255) / 256, 256, 0, stream>>>(pooled, pooled, 128, 0, 32, 32, K33, K33, 2048, btPool);
    wmma_gemm_kernel<<<dim3(2048 / 512, MP / 16), 256, 0, stream>>>(wA2, btPool, K33, o2, OFFC, 1024, 0, nullptr, nullptr, 0);

    // 4. conv3(concat(x,y)) * sigmoid(upsample(o2)) -> outg
    im2col_kernel<<<(8192 * K33 + 255) / 256, 256, 0, stream>>>(x, y, 64, 64, HH, WWD, K33, K33, 8192, btXY);
    wmma_gemm_kernel<<<dim3(8192 / 512, MP / 16), 256, 0, stream>>>(wA3, btXY, K33, outg, OFFC, HW, 1, o2, nullptr, OFFC);

    // 5. conv4(outg) -> offsets
    im2col_kernel<<<(8192 * K44P + 255) / 256, 256, 0, stream>>>(outg, outg, OFFC, 0, HH, WWD, K44R, K44P, 8192, btOut);
    wmma_gemm_kernel<<<dim3(8192 / 512, MP / 16), 256, 0, stream>>>(wA4, btOut, K44P, offs, OFFC, HW, 0, nullptr, nullptr, 0);

    // 6. three deformable conv branches -> fp16 transposed concat buffer
    int dgrid = (BB * CC * HW + 255) / 256;
    deform_kernel<<<dgrid, 256, 0, stream>>>(x, offs, w3, b3, 3, 1, 0,   0, catBT);
    deform_kernel<<<dgrid, 256, 0, stream>>>(x, offs, w5, b5, 5, 2, 18,  64, catBT);
    deform_kernel<<<dgrid, 256, 0, stream>>>(x, offs, w7, b7, 7, 3, 68, 128, catBT);

    // 7. pointwise conv + bias -> d_out
    wmma_gemm_kernel<<<dim3(8192 / 512, 64 / 16), 256, 0, stream>>>(wApw, catBT, KPW, out, 64, HW, 2, nullptr, pw_b, 0);
}
